// GatedRecurrence_70824010711762
// MI455X (gfx1250) — matrix operand order repaired and checked by reading, not compiled, not run on hardware
//
#include <hip/hip_runtime.h>

// ---------------------------------------------------------------------------
// LRU gated-recurrence block for gfx1250 (MI455X), bf16 WMMA GEMMs + f32 scan.
// B=8 S=2048 D=N=1024, M=B*S=16384.
// GEMM: 256-thread blocks (8 waves), block tile 256M x 64N, wave tile 32M x 64N,
// k-step 32, double-buffered LDS staging, 8 v_wmma per wave per k-step.
// Compute holds 2 A-frags and streams B-frags one at a time (low VGPR pressure,
// no scratch spills); staging is interleaved load->store so the scheduler can
// slot the global loads between WMMAs.
// ---------------------------------------------------------------------------

#define Bq   8
#define Sq   2048
#define Dq   1024
#define Nq   1024
#define Mq   (Bq * Sq)            // 16384 token rows

typedef __attribute__((ext_vector_type(16))) __bf16 v16bf;
typedef __attribute__((ext_vector_type(8)))  float  v8f;

union FragBF { uint4 q[2]; v16bf v; };
union AccF   { v8f v; float f[8]; };

static __device__ __forceinline__ unsigned short f32_to_bf16(float f) {
    unsigned int u = __float_as_uint(f);
    u += 0x7FFFu + ((u >> 16) & 1u);      // round-to-nearest-even
    return (unsigned short)(u >> 16);
}
static __device__ __forceinline__ float gelu_exact(float x) {
    return 0.5f * x * (1.0f + erff(x * 0.70710678118654752f));
}
static __device__ __forceinline__ unsigned int pack_bf16x2(float lo, float hi) {
    return (unsigned int)f32_to_bf16(lo) | ((unsigned int)f32_to_bf16(hi) << 16);
}

// ------------------------- parameter kernel --------------------------------
// p[0..N)   = lam_re, p[N..2N) = lam_im, p[2N..3N) = gamma
__global__ void k_params(const float* __restrict__ nu_log,
                         const float* __restrict__ theta_log,
                         float* __restrict__ p) {
    int n = blockIdx.x * blockDim.x + threadIdx.x;
    if (n >= Nq) return;
    float mag   = expf(-expf(nu_log[n]));
    float phase = expf(theta_log[n]);
    p[n]            = mag * cosf(phase);
    p[Nq + n]       = mag * sinf(phase);
    p[2 * Nq + n]   = sqrtf(fmaxf(0.0f, 1.0f - mag * mag));
}

// ------------------------- weight conversion (vectorized x4) ---------------
// 7 segments of N*D (=1M) elems each -> bf16:
// 0: B_re*gamma  1: B_im*gamma  2: C_re  3: C_im  4: D_skip  5: W_gate  6: W_con
__global__ void k_cvt_w(const float* __restrict__ Bre, const float* __restrict__ Bim,
                        const float* __restrict__ Cre, const float* __restrict__ Cim,
                        const float* __restrict__ Dsk, const float* __restrict__ Wg,
                        const float* __restrict__ Wc,  const float* __restrict__ params,
                        unsigned short* __restrict__ wbf) {
    const int QSEG   = (Nq * Dq) / 4;                  // 256K float4 per segment
    const int total4 = 7 * QSEG;
    for (int i = blockIdx.x * blockDim.x + threadIdx.x; i < total4;
         i += gridDim.x * blockDim.x) {
        int seg  = i / QSEG;
        int off4 = i - seg * QSEG;                     // float4 index in segment
        const float* src;
        float scale = 1.0f;
        switch (seg) {
            case 0: src = Bre; scale = params[2 * Nq + ((off4 * 4) >> 10)]; break;
            case 1: src = Bim; scale = params[2 * Nq + ((off4 * 4) >> 10)]; break;
            case 2: src = Cre; break;
            case 3: src = Cim; break;
            case 4: src = Dsk; break;
            case 5: src = Wg;  break;
            default: src = Wc; break;
        }
        float4 f = ((const float4*)src)[off4];
        uint2 o;
        o.x = pack_bf16x2(f.x * scale, f.y * scale);
        o.y = pack_bf16x2(f.z * scale, f.w * scale);
        ((uint2*)wbf)[i] = o;
    }
}

__global__ void k_cvt_x(const float* __restrict__ x, unsigned short* __restrict__ xbf) {
    const int total4 = (Mq * Dq) / 4;
    for (int i = blockIdx.x * blockDim.x + threadIdx.x; i < total4;
         i += gridDim.x * blockDim.x) {
        float4 f = ((const float4*)x)[i];
        uint2 o;
        o.x = pack_bf16x2(f.x, f.y);
        o.y = pack_bf16x2(f.z, f.w);
        ((uint2*)xbf)[i] = o;
    }
}

// ------------------------- bf16 WMMA GEMM ----------------------------------
// Cout[M,Ncols] (f32) (+)= alpha * A[M,K](bf16) * W[Ncols,K](bf16)^T
__global__ __launch_bounds__(256)
void k_gemm_bf16(const unsigned short* __restrict__ A,
                 const unsigned short* __restrict__ W,
                 float* __restrict__ Cout,
                 int Kdim, int Ncols, float alpha, int accumulate) {
    // double-buffered tiles: A 256x32 bf16 (16 KB), B 64x32 bf16 (4 KB)
    __shared__ __align__(16) unsigned short ldsA[2][256 * 32];
    __shared__ __align__(16) unsigned short ldsB[2][64 * 32];

    const int tid    = threadIdx.x;
    const int lane   = tid & 31;
    const int wave   = tid >> 5;
    const int half   = lane >> 4;   // 0: lanes 0-15, 1: lanes 16-31
    const int r16    = lane & 15;
    const int blockM = blockIdx.y * 256;
    const int blockN = blockIdx.x * 64;

    AccF acc[2][4];
#pragma unroll
    for (int i = 0; i < 2; ++i)
#pragma unroll
        for (int j = 0; j < 4; ++j) {
            v8f z = {0.f, 0.f, 0.f, 0.f, 0.f, 0.f, 0.f, 0.f};
            acc[i][j].v = z;
        }

    // ---- stage one k-slab (32 wide) into LDS buffer `buf`; branch-free ----
    auto stage = [&](int buf, int k0) {
#pragma unroll
        for (int c = 0; c < 4; ++c) {
            int idx = tid + c * 256;               // 0..1023 : 256 rows x 4 quads
            int row = idx >> 2, quad = idx & 3;
            const unsigned short* src =
                A + (size_t)(blockM + row) * Kdim + k0 + quad * 8;
            *(uint4*)&ldsA[buf][row * 32 + quad * 8] = *(const uint4*)src;
            __builtin_prefetch(src + 32, 0, 2);    // next k-slab (speculative)
        }
        {
            int row = tid >> 2, quad = tid & 3;    // 64 rows x 4 quads
            const unsigned short* src =
                W + (size_t)(blockN + row) * Kdim + k0 + quad * 8;
            *(uint4*)&ldsB[buf][row * 32 + quad * 8] = *(const uint4*)src;
            __builtin_prefetch(src + 32, 0, 2);
        }
    };

    // ---- 8 WMMAs from buffer `buf`: hold 2 A-frags, stream B-frags ----
    auto compute = [&](int buf) {
        FragBF a[2];
#pragma unroll
        for (int i = 0; i < 2; ++i) {
            const unsigned short* la =
                &ldsA[buf][(wave * 32 + i * 16 + r16) * 32 + half * 8];
            a[i].q[0] = *(const uint4*)la;          // K = 0..7   / 8..15
            a[i].q[1] = *(const uint4*)(la + 16);   // K = 16..23 / 24..31
        }
#pragma unroll
        for (int j = 0; j < 4; ++j) {
            FragBF b;
            const unsigned short* lb = &ldsB[buf][(j * 16 + r16) * 32 + half * 8];
            b.q[0] = *(const uint4*)lb;             // K = 0..7   / 8..15
            b.q[1] = *(const uint4*)(lb + 16);      // K = 16..23 / 24..31
            acc[0][j].v = __builtin_amdgcn_wmma_f32_16x16x32_bf16(
                false, a[0].v, false, b.v, (short)0, acc[0][j].v, false, false);
            acc[1][j].v = __builtin_amdgcn_wmma_f32_16x16x32_bf16(
                false, a[1].v, false, b.v, (short)0, acc[1][j].v, false, false);
        }
    };

    stage(0, 0);
    __syncthreads();
    int buf = 0;
    for (int k0 = 32; k0 < Kdim; k0 += 32) {
        stage(buf ^ 1, k0);     // fill next buffer (loads slot between WMMAs)
        compute(buf);           // consume current buffer
        __syncthreads();
        buf ^= 1;
    }
    compute(buf);

    // ---- epilogue: C/D layout VGPR r -> M=r (lanes 0-15) / M=8+r (lanes 16-31)
    const int rowBase = blockM + wave * 32 + half * 8;
#pragma unroll
    for (int i = 0; i < 2; ++i)
#pragma unroll
        for (int j = 0; j < 4; ++j) {
            int col = blockN + j * 16 + r16;
#pragma unroll
            for (int r = 0; r < 8; ++r) {
                size_t idx = (size_t)(rowBase + i * 16 + r) * Ncols + col;
                float v = alpha * acc[i][j].f[r];
                if (accumulate) v += Cout[idx];
                Cout[idx] = v;
            }
        }
}

// ------------------------- sequential complex scan -------------------------
__global__ void k_scan(const float* __restrict__ ur, const float* __restrict__ ui,
                       const float* __restrict__ params,
                       const float* __restrict__ h0r, const float* __restrict__ h0i,
                       unsigned short* __restrict__ hsr, unsigned short* __restrict__ hsi,
                       float* __restrict__ hfr, float* __restrict__ hfi) {
    int j = blockIdx.x * blockDim.x + threadIdx.x;   // 0 .. B*N-1
    if (j >= Bq * Nq) return;
    int b = j >> 10;
    int n = j & (Nq - 1);
    float lr = params[n], li = params[Nq + n];
    float hr = h0r[j],    hi = h0i[j];
    size_t base = (size_t)b * Sq * Nq + n;
    for (int t = 0; t < Sq; ++t) {
        size_t idx = base + (size_t)t * Nq;
        float nr = fmaf(lr, hr, fmaf(-li, hi, ur[idx]));
        float ni = fmaf(lr, hi, fmaf( li, hr, ui[idx]));
        hr = nr; hi = ni;
        hsr[idx] = f32_to_bf16(nr);
        hsi[idx] = f32_to_bf16(ni);
    }
    hfr[j] = hr;
    hfi[j] = hi;
}

// ------------------------- fused LN/gate/GELU ------------------------------
static __device__ __forceinline__ void reduce2(float* red, float& s, float& ss) {
    int tid = threadIdx.x;
    red[tid] = s; red[256 + tid] = ss;
    __syncthreads();
    for (int off = 128; off > 0; off >>= 1) {
        if (tid < off) { red[tid] += red[tid + off]; red[256 + tid] += red[256 + tid + off]; }
        __syncthreads();
    }
    s = red[0]; ss = red[256];
    __syncthreads();
}

// one block per row m; y_bf = bf16( gelu(ln( ln(y_raw) * gelu(ln(g_lin)) )) )
__global__ __launch_bounds__(256)
void k_fuse(const float* __restrict__ yraw, const float* __restrict__ glin,
            unsigned short* __restrict__ ybf) {
    __shared__ float red[512];
    const float invN = 1.0f / (float)Nq;
    int tid = threadIdx.x;
    size_t rb = (size_t)blockIdx.x * Nq;

    float ry[4], rg[4];
    float s = 0.f, ss = 0.f, gs = 0.f, gss = 0.f;
#pragma unroll
    for (int i = 0; i < 4; ++i) {
        int idx = tid + 256 * i;
        ry[i] = yraw[rb + idx];
        rg[i] = glin[rb + idx];
        s  += ry[i]; ss  += ry[i] * ry[i];
        gs += rg[i]; gss += rg[i] * rg[i];
    }
    reduce2(red, s, ss);
    float mu  = s * invN;
    float inv = rsqrtf(fmaxf(0.f, ss * invN - mu * mu) + 1e-5f);
    reduce2(red, gs, gss);
    float gmu  = gs * invN;
    float ginv = rsqrtf(fmaxf(0.f, gss * invN - gmu * gmu) + 1e-5f);

    float z[4], zs = 0.f, zss = 0.f;
#pragma unroll
    for (int i = 0; i < 4; ++i) {
        float ln_y = (ry[i] - mu)  * inv;
        float gate = gelu_exact((rg[i] - gmu) * ginv);
        z[i] = ln_y * gate;
        zs += z[i]; zss += z[i] * z[i];
    }
    reduce2(red, zs, zss);
    float zmu  = zs * invN;
    float zinv = rsqrtf(fmaxf(0.f, zss * invN - zmu * zmu) + 1e-5f);
#pragma unroll
    for (int i = 0; i < 4; ++i) {
        int idx = tid + 256 * i;
        ybf[rb + idx] = f32_to_bf16(gelu_exact((z[i] - zmu) * zinv));
    }
}

// ---------------------------------------------------------------------------
extern "C" void kernel_launch(void* const* d_in, const int* in_sizes, int n_in,
                              void* d_out, int out_size, void* d_ws, size_t ws_size,
                              hipStream_t stream) {
    const float* x      = (const float*)d_in[0];
    const float* h0r    = (const float*)d_in[1];
    const float* h0i    = (const float*)d_in[2];
    const float* nu     = (const float*)d_in[3];
    const float* th     = (const float*)d_in[4];
    const float* Bre    = (const float*)d_in[5];
    const float* Bim    = (const float*)d_in[6];
    const float* Cre    = (const float*)d_in[7];
    const float* Cim    = (const float*)d_in[8];
    const float* Dsk    = (const float*)d_in[9];
    const float* Wg     = (const float*)d_in[10];
    const float* Wc     = (const float*)d_in[11];
    float* out = (float*)d_out;                       // [M,D] ++ hf_re ++ hf_im

    // ---- workspace layout (bytes) ----
    char* ws = (char*)d_ws;
    const size_t SEG = (size_t)Nq * Dq;               // 1M elems
    float*          params = (float*)(ws);                              // 12 KB
    unsigned short* wbf    = (unsigned short*)(ws + 16384);             // 7 * 2 MB
    unsigned short* xbf    = (unsigned short*)(ws + 16384 + 7 * SEG * 2);   // 32 MB (reused for y_bf16)
    float*          bufA   = (float*)((char*)xbf + (size_t)Mq * Dq * 2);    // 64 MB: u_re -> y_raw
    float*          bufB   = (float*)((char*)bufA + (size_t)Mq * Nq * 4);   // 64 MB: u_im -> g_lin
    unsigned short* hsre   = (unsigned short*)((char*)bufB + (size_t)Mq * Nq * 4); // 32 MB
    unsigned short* hsim   = hsre + (size_t)Mq * Nq;                         // 32 MB

    float* hfr = out + (size_t)Mq * Dq;
    float* hfi = hfr + Bq * Nq;

    dim3 gemmGrid(Nq / 64, Mq / 256);                 // (16, 64)
    const int TB = 256;

    // 1. lambda / gamma
    k_params<<<(Nq + TB - 1) / TB, TB, 0, stream>>>(nu, th, params);
    // 2. weights -> bf16 (gamma folded into B_re/B_im rows)
    k_cvt_w<<<2048, TB, 0, stream>>>(Bre, Bim, Cre, Cim, Dsk, Wg, Wc, params, wbf);
    // 3. x -> bf16
    k_cvt_x<<<4096, TB, 0, stream>>>(x, xbf);
    // 4/5. u_re, u_im = (x * B^T) * gamma   (gamma pre-folded)
    k_gemm_bf16<<<gemmGrid, TB, 0, stream>>>(xbf, wbf + 0 * SEG, bufA, Dq, Nq, 1.0f, 0);
    k_gemm_bf16<<<gemmGrid, TB, 0, stream>>>(xbf, wbf + 1 * SEG, bufB, Dq, Nq, 1.0f, 0);
    // 6. sequential complex scan -> bf16 hs, final hidden -> d_out tail
    k_scan<<<(Bq * Nq) / TB, TB, 0, stream>>>(bufA, bufB, params, h0r, h0i,
                                              hsre, hsim, hfr, hfi);
    // 7-9. y_raw = x*D_skip^T + hs_re*C_re^T - hs_im*C_im^T   (bufA reused)
    k_gemm_bf16<<<gemmGrid, TB, 0, stream>>>(xbf,  wbf + 4 * SEG, bufA, Dq, Nq,  1.0f, 0);
    k_gemm_bf16<<<gemmGrid, TB, 0, stream>>>(hsre, wbf + 2 * SEG, bufA, Nq, Nq,  1.0f, 1);
    k_gemm_bf16<<<gemmGrid, TB, 0, stream>>>(hsim, wbf + 3 * SEG, bufA, Nq, Nq, -1.0f, 1);
    // 10. gate pre-activation  (bufB reused)
    k_gemm_bf16<<<gemmGrid, TB, 0, stream>>>(xbf, wbf + 5 * SEG, bufB, Dq, Nq, 1.0f, 0);
    // 11. fused LN -> gate -> LN -> GELU, write bf16 y into xbf (x no longer needed)
    k_fuse<<<Mq, TB, 0, stream>>>(bufA, bufB, xbf);
    // 12. out = y * W_con^T
    k_gemm_bf16<<<gemmGrid, TB, 0, stream>>>(xbf, wbf + 6 * SEG, out, Nq, Dq, 1.0f, 0);
}
